// QuantAwareAttention_65687229825559
// MI455X (gfx1250) — matrix operand order repaired and checked by reading, not compiled, not run on hardware
//
#include <hip/hip_runtime.h>

typedef unsigned short u16;
typedef unsigned int   u32;

typedef __attribute__((ext_vector_type(16))) __bf16 v16bf;
typedef __attribute__((ext_vector_type(8)))  float  v8f;
typedef __attribute__((ext_vector_type(4)))  unsigned int v4u;
typedef __attribute__((ext_vector_type(8)))  int v8i;
typedef __attribute__((ext_vector_type(4)))  int v4i;

#define USE_TDM 1

// ---------------------------------------------------------------- helpers
__device__ __forceinline__ u16 f2bf(float f) {
  u32 x = __float_as_uint(f);
  u32 r = x + 0x7FFFu + ((x >> 16) & 1u);   // round-to-nearest-even
  return (u16)(r >> 16);
}

__device__ __forceinline__ void wait_tensorcnt0() {
#if __has_builtin(__builtin_amdgcn_s_wait_tensorcnt)
  __builtin_amdgcn_s_wait_tensorcnt(0);
#else
  asm volatile("s_wait_tensorcnt 0x0" ::: "memory");
#endif
}
__device__ __forceinline__ void wait_tensorcnt1() {
#if __has_builtin(__builtin_amdgcn_s_wait_tensorcnt)
  __builtin_amdgcn_s_wait_tensorcnt(1);
#else
  asm volatile("s_wait_tensorcnt 0x1" ::: "memory");
#endif
}

union FragBF { v16bf v; uint4 q[2]; u16 s[16]; };

__device__ __forceinline__ float quant_sym(float v, float scale) {
  return fminf(fmaxf(rintf(v / scale), -128.f), 127.f) * scale;
}

// ---------------------------------------------------------------- tiny kernels
__global__ void init_amax_kernel(u32* slots) {
  if (threadIdx.x < 8) slots[threadIdx.x] = 0u;
}

// x f32 -> y bf16, 4 elems/thread
__global__ __launch_bounds__(256) void cvt_f32_bf16_kernel(
    const float* __restrict__ x, u16* __restrict__ y) {
  size_t i = ((size_t)blockIdx.x * 256 + threadIdx.x) * 4;
  float4 v = *(const float4*)(x + i);
  ushort4 o;
  o.x = f2bf(v.x); o.y = f2bf(v.y); o.z = f2bf(v.z); o.w = f2bf(v.w);
  *(ushort4*)(y + i) = o;
}

// per-row (per output channel) symmetric fake-quant of a [N,K] weight -> bf16
__global__ __launch_bounds__(256) void quant_weight_kernel(
    const float* __restrict__ w, u16* __restrict__ wq, int Kdim) {
  __shared__ float red[8];
  __shared__ float s_scale;
  const int row = blockIdx.x;
  const float* wr = w + (size_t)row * Kdim;
  float m = 0.f;
  for (int i = threadIdx.x; i < Kdim; i += 256) m = fmaxf(m, fabsf(wr[i]));
#pragma unroll
  for (int s = 16; s >= 1; s >>= 1) m = fmaxf(m, __shfl_xor(m, s, 32));
  if ((threadIdx.x & 31) == 0) red[threadIdx.x >> 5] = m;
  __syncthreads();
  if (threadIdx.x == 0) {
    float t = red[0];
#pragma unroll
    for (int i = 1; i < 8; ++i) t = fmaxf(t, red[i]);
    s_scale = fmaxf(t / 127.0f, 1e-8f);
  }
  __syncthreads();
  const float scale = s_scale;
  for (int i = threadIdx.x; i < Kdim; i += 256)
    wq[(size_t)row * Kdim + i] = f2bf(quant_sym(wr[i], scale));
}

// per-tensor fake-quant in-place (f32), fused amax of the quantized tensor
__global__ __launch_bounds__(256) void quant_f32_amax_kernel(
    float* __restrict__ x, const u32* __restrict__ slot_in,
    u32* __restrict__ slot_out) {
  __shared__ float red[8];
  size_t i = ((size_t)blockIdx.x * 256 + threadIdx.x) * 4;
  const float scale = fmaxf(__uint_as_float(*slot_in) / 127.0f, 1e-8f);
  float4 v = *(const float4*)(x + i);
  v.x = quant_sym(v.x, scale); v.y = quant_sym(v.y, scale);
  v.z = quant_sym(v.z, scale); v.w = quant_sym(v.w, scale);
  *(float4*)(x + i) = v;
  float m = fmaxf(fmaxf(fabsf(v.x), fabsf(v.y)), fmaxf(fabsf(v.z), fabsf(v.w)));
#pragma unroll
  for (int s = 16; s >= 1; s >>= 1) m = fmaxf(m, __shfl_xor(m, s, 32));
  if ((threadIdx.x & 31) == 0) red[threadIdx.x >> 5] = m;
  __syncthreads();
  if (threadIdx.x == 0) {
    float t = red[0];
#pragma unroll
    for (int j = 1; j < 8; ++j) t = fmaxf(t, red[j]);
    atomicMax(slot_out, __float_as_uint(t));
  }
}

// per-tensor fake-quant, f32 -> bf16, 4 elems/thread
__global__ __launch_bounds__(256) void quant_bf16_kernel(
    const float* __restrict__ x, u16* __restrict__ y,
    const u32* __restrict__ slot) {
  size_t i = ((size_t)blockIdx.x * 256 + threadIdx.x) * 4;
  const float scale = fmaxf(__uint_as_float(*slot) / 127.0f, 1e-8f);
  float4 v = *(const float4*)(x + i);
  ushort4 o;
  o.x = f2bf(quant_sym(v.x, scale)); o.y = f2bf(quant_sym(v.y, scale));
  o.z = f2bf(quant_sym(v.z, scale)); o.w = f2bf(quant_sym(v.w, scale));
  *(ushort4*)(y + i) = o;
}

// V: per-tensor fake-quant (f32 in) + transpose to [B*H, DH, T] bf16
__global__ void quant_transpose_v_kernel(
    const float* __restrict__ in, u16* __restrict__ out,
    const u32* __restrict__ slot, int T_, int H_, int DH_) {
  __shared__ u16 tile[32][40];
  const int bh = blockIdx.z, b = bh / H_, h = bh % H_;
  const int t0 = blockIdx.x * 32, d0 = blockIdx.y * 32;
  const int tx = threadIdx.x, ty = threadIdx.y;   // 32 x 8
  const int Dfull = H_ * DH_;
  const float scale = fmaxf(__uint_as_float(*slot) / 127.0f, 1e-8f);
#pragma unroll
  for (int j = 0; j < 32; j += 8) {
    float vv = in[((size_t)b * T_ + t0 + ty + j) * Dfull + h * DH_ + d0 + tx];
    tile[ty + j][tx] = f2bf(quant_sym(vv, scale));
  }
  __syncthreads();
#pragma unroll
  for (int j = 0; j < 32; j += 8)
    out[((size_t)bh * DH_ + d0 + ty + j) * T_ + t0 + tx] = tile[tx][ty + j];
}

// ---------------------------------------------------------------- bf16 WMMA GEMM
//   C[M,N] f32 = A[M,K](bf16) * W[N,K](bf16)^T + bias[N], optional fused amax
// block tile 128x128, 8 waves (4x2), wave tile 32x64, K-step 32, double-buffered.
#define GEMM_LDA 40   // padded LDS row stride (u16 elems): rows 16B-aligned

__global__ __launch_bounds__(256) void gemm_bf16_kernel(
    const u16* __restrict__ A, const u16* __restrict__ W,
    const float* __restrict__ bias, float* __restrict__ C,
    int M, int N, int K, u32* amax_slot) {
  __shared__ __align__(16) u16 As[2][128 * GEMM_LDA];
  __shared__ __align__(16) u16 Bs[2][128 * GEMM_LDA];
  __shared__ float redmax[8];

  const int tid  = threadIdx.x;
  const int lane = tid & 31;
  const int wave = tid >> 5;
  const int wm   = wave & 3;
  const int wn   = wave >> 2;
  const int l16  = lane & 15;
  const int lh   = lane >> 4;
  const int bm   = blockIdx.x * 128;
  const int bn   = blockIdx.y * 128;

  const int r0 = tid >> 1;           // staging row 0..127
  const int c0 = (tid & 1) * 16;     // staging col 0 or 16

  const v8f vzero = {0.f, 0.f, 0.f, 0.f, 0.f, 0.f, 0.f, 0.f};
  v8f acc[2][4];
#pragma unroll
  for (int mi = 0; mi < 2; ++mi)
#pragma unroll
    for (int ni = 0; ni < 4; ++ni) acc[mi][ni] = vzero;

  uint4 ra0, ra1, rb0, rb1;
  const u16* pa = &A[(size_t)(bm + r0) * K + c0];
  const u16* pb = &W[(size_t)(bn + r0) * K + c0];

  ra0 = *(const uint4*)pa;       ra1 = *(const uint4*)(pa + 8);
  rb0 = *(const uint4*)pb;       rb1 = *(const uint4*)(pb + 8);
  *(uint4*)&As[0][r0 * GEMM_LDA + c0]     = ra0;
  *(uint4*)&As[0][r0 * GEMM_LDA + c0 + 8] = ra1;
  *(uint4*)&Bs[0][r0 * GEMM_LDA + c0]     = rb0;
  *(uint4*)&Bs[0][r0 * GEMM_LDA + c0 + 8] = rb1;

  int cur = 0;
  for (int k0 = 0; k0 < K; k0 += 32) {
    __syncthreads();
    const bool more = (k0 + 32) < K;
    if (more) {
      ra0 = *(const uint4*)(pa + k0 + 32);
      ra1 = *(const uint4*)(pa + k0 + 40);
      rb0 = *(const uint4*)(pb + k0 + 32);
      rb1 = *(const uint4*)(pb + k0 + 40);
      if (k0 + 64 < K) {
        __builtin_prefetch(pa + k0 + 64, 0, 1);
        __builtin_prefetch(pb + k0 + 64, 0, 1);
      }
    }

    FragBF af[2], bf[4];
#pragma unroll
    for (int mi = 0; mi < 2; ++mi) {
      int r = wm * 32 + mi * 16 + l16;
      af[mi].q[0] = *(const uint4*)&As[cur][r * GEMM_LDA + lh * 8];
      af[mi].q[1] = *(const uint4*)&As[cur][r * GEMM_LDA + lh * 8 + 16];
    }
#pragma unroll
    for (int ni = 0; ni < 4; ++ni) {
      int r = wn * 64 + ni * 16 + l16;
      bf[ni].q[0] = *(const uint4*)&Bs[cur][r * GEMM_LDA + lh * 8];
      bf[ni].q[1] = *(const uint4*)&Bs[cur][r * GEMM_LDA + lh * 8 + 16];
    }
#pragma unroll
    for (int mi = 0; mi < 2; ++mi)
#pragma unroll
      for (int ni = 0; ni < 4; ++ni)
        acc[mi][ni] = __builtin_amdgcn_wmma_f32_16x16x32_bf16(
            false, af[mi].v, false, bf[ni].v, (short)0, acc[mi][ni], false, false);

    if (more) {
      int nb = cur ^ 1;
      *(uint4*)&As[nb][r0 * GEMM_LDA + c0]     = ra0;
      *(uint4*)&As[nb][r0 * GEMM_LDA + c0 + 8] = ra1;
      *(uint4*)&Bs[nb][r0 * GEMM_LDA + c0]     = rb0;
      *(uint4*)&Bs[nb][r0 * GEMM_LDA + c0 + 8] = rb1;
    }
    cur ^= 1;
  }

  // epilogue: bias add, store f32, fused local amax
  float mloc = 0.f;
#pragma unroll
  for (int mi = 0; mi < 2; ++mi) {
#pragma unroll
    for (int ni = 0; ni < 4; ++ni) {
      int n = bn + wn * 64 + ni * 16 + l16;
      float bv = bias[n];
#pragma unroll
      for (int r = 0; r < 8; ++r) {
        int m = bm + wm * 32 + mi * 16 + lh * 8 + r;
        float v = acc[mi][ni][r] + bv;
        mloc = fmaxf(mloc, fabsf(v));
        C[(size_t)m * N + n] = v;
      }
    }
  }
  if (amax_slot) {
#pragma unroll
    for (int s = 16; s >= 1; s >>= 1) mloc = fmaxf(mloc, __shfl_xor(mloc, s, 32));
    if (lane == 0) redmax[wave] = mloc;
    __syncthreads();
    if (tid == 0) {
      float t = redmax[0];
#pragma unroll
      for (int j = 1; j < 8; ++j) t = fmaxf(t, redmax[j]);
      atomicMax(amax_slot, __float_as_uint(t));
    }
  }
}

// ---------------------------------------------------------------- flash attention (bf16 WMMA)
// grid (T/128, B*H), block 256 (8 waves); each wave owns 16 q rows.
// K tiles (64x128 bf16) staged into LDS by the Tensor Data Mover, double-buffered.
__global__ __launch_bounds__(256) void attn_kernel(
    const u16* __restrict__ qb, const u16* __restrict__ kb,
    const u16* __restrict__ vt, u16* __restrict__ ctx,
    int Tc, int Hc, int DHc) {
  __shared__ __align__(16) u16 Ks[2][64 * 128];   // 2 x 16 KB K tiles
  __shared__ __align__(16) u16 Pl[8][16 * 64];    // per-wave P strips

  const int tid  = threadIdx.x;
  const int lane = tid & 31;
  const int wave = tid >> 5;
  const int l16  = lane & 15;
  const int lh   = lane >> 4;
  const int qt   = blockIdx.x * 128;
  const int bh   = blockIdx.y;
  const int b    = bh / Hc;
  const int h    = bh % Hc;
  const int Dful = Hc * DHc;
  const float sm_scale = rsqrtf((float)DHc);

  const size_t baseBH = (size_t)b * Tc * Dful + (size_t)h * DHc;
  const u16* Kbase = kb + baseBH;

#if USE_TDM
  union G1 { v8i v; u16 h[16]; u32 w[8]; } g1;
#pragma unroll
  for (int i = 0; i < 8; ++i) g1.w[i] = 0u;
  g1.h[1]  = 1;                       // data_size = 2 bytes
  g1.h[3]  = (u16)(Dful & 0xFFFF);    // tensor_dim0
  g1.h[4]  = (u16)(Dful >> 16);
  g1.h[5]  = (u16)(Tc & 0xFFFF);      // tensor_dim1
  g1.h[6]  = (u16)(Tc >> 16);
  g1.h[7]  = (u16)DHc;                // tile_dim0 = 128 elems per row
  g1.h[8]  = 64;                      // tile_dim1 = 64 rows
  g1.h[10] = (u16)(Dful & 0xFFFF);    // tensor_dim0_stride (48b)
  g1.h[11] = (u16)(Dful >> 16);
  union GZ { v4i v; u32 w[4]; } gz;
#pragma unroll
  for (int i = 0; i < 4; ++i) gz.w[i] = 0;

  auto issue_tdm = [&](int it, int buf) {
    unsigned long long ga =
        (unsigned long long)(Kbase + (size_t)(it * 64) * Dful);
    union { v4u v; u32 w[4]; } g0;
    g0.w[0] = 1u;                                   // count=1, user descriptor
    g0.w[1] = (u32)(unsigned long long)&Ks[buf][0]; // lds_addr (bytes)
    g0.w[2] = (u32)ga;                              // global_addr[31:0]
    g0.w[3] = (u32)(ga >> 32) | (2u << 30);         // addr[56:32] | type=2
    asm volatile("tensor_load_to_lds %0, %1, %2, %3"
                 :
                 : "s"(g0.v), "s"(g1.v), "s"(gz.v), "s"(gz.v)
                 : "memory");
  };
#endif

  FragBF qf[4];
  {
    const u16* Qp = qb + baseBH + (size_t)(qt + wave * 16 + l16) * Dful;
#pragma unroll
    for (int ks = 0; ks < 4; ++ks) {
      int c = ks * 32 + lh * 8;
      qf[ks].q[0] = *(const uint4*)(Qp + c);
      qf[ks].q[1] = *(const uint4*)(Qp + c + 16);
    }
  }

  const v8f vzero = {0.f, 0.f, 0.f, 0.f, 0.f, 0.f, 0.f, 0.f};
  v8f o[8];
#pragma unroll
  for (int nt = 0; nt < 8; ++nt) o[nt] = vzero;
  float mrow[8], lrow[8];
#pragma unroll
  for (int r = 0; r < 8; ++r) { mrow[r] = -1e30f; lrow[r] = 0.f; }

  const int nkt = Tc / 64;
#if USE_TDM
  if (wave == 0) issue_tdm(0, 0);
#endif

  for (int it = 0; it < nkt; ++it) {
    const int cur = it & 1;
    const int kt = it * 64;

#if USE_TDM
    __syncthreads();                   // prior readers of Ks[cur^1] done
    if (wave == 0) {
      if (it + 1 < nkt) { issue_tdm(it + 1, cur ^ 1); wait_tensorcnt1(); }
      else              { wait_tensorcnt0(); }
    }
    __syncthreads();                   // Ks[cur] visible to all waves
#else
    __syncthreads();
    for (int c = tid; c < (64 * 128) / 8; c += 256) {
      int row = c >> 4, col = (c & 15) * 8;
      *(uint4*)&Ks[cur][row * 128 + col] =
          *(const uint4*)&Kbase[(size_t)(kt + row) * Dful + col];
    }
    __syncthreads();
#endif

    // ---- S = Q K^T (16x64 f32)
    v8f s[4];
#pragma unroll
    for (int ni = 0; ni < 4; ++ni) s[ni] = vzero;
#pragma unroll
    for (int ni = 0; ni < 4; ++ni) {
      const u16* Kr = &Ks[cur][(ni * 16 + l16) * 128];
#pragma unroll
      for (int ks = 0; ks < 4; ++ks) {
        FragBF kf;
        kf.q[0] = *(const uint4*)(Kr + ks * 32 + lh * 8);
        kf.q[1] = *(const uint4*)(Kr + ks * 32 + lh * 8 + 16);
        s[ni] = __builtin_amdgcn_wmma_f32_16x16x32_bf16(
            false, qf[ks].v, false, kf.v, (short)0, s[ni], false, false);
      }
    }

    // ---- online softmax
    float cmax[8];
#pragma unroll
    for (int r = 0; r < 8; ++r) cmax[r] = -1e30f;
#pragma unroll
    for (int ni = 0; ni < 4; ++ni)
#pragma unroll
      for (int r = 0; r < 8; ++r) cmax[r] = fmaxf(cmax[r], s[ni][r] * sm_scale);
#pragma unroll
    for (int r = 0; r < 8; ++r) {
#pragma unroll
      for (int msk = 1; msk <= 8; msk <<= 1)
        cmax[r] = fmaxf(cmax[r], __shfl_xor(cmax[r], msk, 32));
    }
    float fac[8], rsum[8];
#pragma unroll
    for (int r = 0; r < 8; ++r) {
      float mn = fmaxf(mrow[r], cmax[r]);
      fac[r]  = __expf(mrow[r] - mn);
      mrow[r] = mn;
      rsum[r] = 0.f;
    }
#pragma unroll
    for (int ni = 0; ni < 4; ++ni)
#pragma unroll
      for (int r = 0; r < 8; ++r) {
        float p = __expf(s[ni][r] * sm_scale - mrow[r]);
        rsum[r] += p;
        Pl[wave][(lh * 8 + r) * 64 + ni * 16 + l16] = f2bf(p);
      }
#pragma unroll
    for (int r = 0; r < 8; ++r) {
#pragma unroll
      for (int msk = 1; msk <= 8; msk <<= 1)
        rsum[r] += __shfl_xor(rsum[r], msk, 32);
      lrow[r] = lrow[r] * fac[r] + rsum[r];
    }
#pragma unroll
    for (int nt = 0; nt < 8; ++nt)
#pragma unroll
      for (int r = 0; r < 8; ++r) o[nt][r] *= fac[r];

    // ---- O += P V (A = P via LDS, B = V^T fragments from global)
#pragma unroll
    for (int ks2 = 0; ks2 < 2; ++ks2) {
      FragBF pf;
      pf.q[0] = *(const uint4*)&Pl[wave][l16 * 64 + ks2 * 32 + lh * 8];
      pf.q[1] = *(const uint4*)&Pl[wave][l16 * 64 + ks2 * 32 + lh * 8 + 16];
#pragma unroll
      for (int nt = 0; nt < 8; ++nt) {
        const u16* Vp = vt + ((size_t)bh * DHc + nt * 16 + l16) * Tc
                           + kt + ks2 * 32 + lh * 8;
        FragBF vf;
        vf.q[0] = *(const uint4*)Vp;
        vf.q[1] = *(const uint4*)(Vp + 16);
        o[nt] = __builtin_amdgcn_wmma_f32_16x16x32_bf16(
            false, pf.v, false, vf.v, (short)0, o[nt], false, false);
      }
    }
  }

#pragma unroll
  for (int r = 0; r < 8; ++r) {
    float invl = 1.0f / lrow[r];
    int t = qt + wave * 16 + lh * 8 + r;
#pragma unroll
    for (int nt = 0; nt < 8; ++nt)
      ctx[((size_t)b * Tc + t) * Dful + h * DHc + nt * 16 + l16] =
          f2bf(o[nt][r] * invl);
  }
}

// ---------------------------------------------------------------- launcher
extern "C" void kernel_launch(void* const* d_in, const int* in_sizes, int n_in,
                              void* d_out, int out_size, void* d_ws, size_t ws_size,
                              hipStream_t stream) {
  (void)in_sizes; (void)n_in; (void)out_size; (void)ws_size;
  constexpr int Bc = 2, Tc = 2048, Dc = 2048, Hc = 16, DHc = 128;
  constexpr size_t SZX = (size_t)Bc * Tc * Dc;   // 8,388,608 activations
  constexpr size_t SZW = (size_t)Dc * Dc;        // 4,194,304 weight elems
  constexpr int Mrows = Bc * Tc;                 // 4096

  const float* x  = (const float*)d_in[0];
  const float* wq = (const float*)d_in[1];
  const float* bq = (const float*)d_in[2];
  const float* wk = (const float*)d_in[3];
  const float* bk = (const float*)d_in[4];
  const float* wv = (const float*)d_in[5];
  const float* bv = (const float*)d_in[6];
  const float* wo = (const float*)d_in[7];
  const float* bo = (const float*)d_in[8];
  float* out = (float*)d_out;

  char* ws = (char*)d_ws;
  u32* amax = (u32*)ws;                      // slots: 0=q 1=k1 2=v1 3=k2 4=v2
  u16* xb   = (u16*)(ws + 256);
  u16* wqb  = xb  + SZX;
  u16* wkb  = wqb + SZW;
  u16* wvb  = wkb + SZW;
  u16* wob  = wvb + SZW;
  float* qf = (float*)(wob + SZW);
  float* kf = qf + SZX;
  float* vf = kf + SZX;
  u16* qb   = (u16*)(vf + SZX);
  u16* kb   = qb + SZX;
  u16* vt   = kb + SZX;
  u16* ctx  = vt + SZX;

  const int EB = 256;
  const int nV4 = (int)(SZX / (EB * 4));     // 8192 blocks, 4 elems/thread

  init_amax_kernel<<<1, 32, 0, stream>>>(amax);
  cvt_f32_bf16_kernel<<<nV4, EB, 0, stream>>>(x, xb);

  quant_weight_kernel<<<Dc, EB, 0, stream>>>(wq, wqb, Dc);
  quant_weight_kernel<<<Dc, EB, 0, stream>>>(wk, wkb, Dc);
  quant_weight_kernel<<<Dc, EB, 0, stream>>>(wv, wvb, Dc);
  quant_weight_kernel<<<Dc, EB, 0, stream>>>(wo, wob, Dc);

  dim3 gg(Mrows / 128, Dc / 128);
  gemm_bf16_kernel<<<gg, 256, 0, stream>>>(xb, wqb, bq, qf, Mrows, Dc, Dc, &amax[0]);
  gemm_bf16_kernel<<<gg, 256, 0, stream>>>(xb, wkb, bk, kf, Mrows, Dc, Dc, &amax[1]);
  gemm_bf16_kernel<<<gg, 256, 0, stream>>>(xb, wvb, bv, vf, Mrows, Dc, Dc, &amax[2]);

  // q: single per-tensor fake-quant -> bf16
  quant_bf16_kernel<<<nV4, EB, 0, stream>>>(qf, qb, &amax[0]);
  // k: fake-quant (ABITS) in f32 + fused amax, then fake-quant (KVBITS) -> bf16
  quant_f32_amax_kernel<<<nV4, EB, 0, stream>>>(kf, &amax[1], &amax[3]);
  quant_bf16_kernel<<<nV4, EB, 0, stream>>>(kf, kb, &amax[3]);
  // v: fake-quant + fused amax, then fused quant+transpose -> [B*H, DH, T]
  quant_f32_amax_kernel<<<nV4, EB, 0, stream>>>(vf, &amax[2], &amax[4]);
  dim3 gt(Tc / 32, DHc / 32, Bc * Hc);
  quant_transpose_v_kernel<<<gt, dim3(32, 8), 0, stream>>>(vf, vt, &amax[4], Tc, Hc, DHc);

  dim3 ga(Tc / 128, Bc * Hc);
  attn_kernel<<<ga, 256, 0, stream>>>(qb, kb, vt, ctx, Tc, Hc, DHc);

  gemm_bf16_kernel<<<gg, 256, 0, stream>>>(ctx, wob, bo, out, Mrows, Dc, Dc, nullptr);
}
